// AttentionDecoder_20186346292015
// MI455X (gfx1250) — matrix operand order repaired and checked by reading, not compiled, not run on hardware
//
#include <hip/hip_runtime.h>
#include <math.h>
#include <stdint.h>

// ---------------- problem dimensions ----------------
// B=64, S=1024, E=512, H=512, HKQ=512, V=256, T=400, SOS=1
#define NBLK 64          // persistent decoder blocks (conservative residency)
#define NTHR 256         // 8 waves/block -> 512 waves total
#define NTID (NBLK*NTHR)

typedef __attribute__((ext_vector_type(16))) _Float16 v16h;
typedef __attribute__((ext_vector_type(8)))  float    v8f;
typedef __attribute__((ext_vector_type(4)))  unsigned v4u;
typedef __attribute__((ext_vector_type(8)))  int      v8i;
typedef __attribute__((ext_vector_type(4)))  int      v4i;

#define DEV static __device__ __forceinline__

// ---------------- workspace layout (bytes, 256-aligned) ----------------
constexpr size_t OFF_BAR   = 0;                      // barrier counters (256 B)
constexpr size_t OFF_H1    = 256;                    // fp32 [64][512]
constexpr size_t OFF_C1    = OFF_H1  + 131072;
constexpr size_t OFF_H2    = OFF_C1  + 131072;
constexpr size_t OFF_C2    = OFF_H2  + 131072;
constexpr size_t OFF_CTX   = OFF_C2  + 131072;
constexpr size_t OFF_XCAT1 = OFF_CTX + 131072;       // f16 [64][1536] = [x|ctx|h1]
constexpr size_t OFF_XCAT2 = OFF_XCAT1 + 196608;     // f16 [64][1024] = [h1|h2]
constexpr size_t OFF_ZIN   = OFF_XCAT2 + 131072;     // f16 [64][1024] = [h2|ctx]
constexpr size_t OFF_ZF    = OFF_ZIN   + 131072;     // f16 [64][512]  = relu proj
constexpr size_t OFF_ZERO_END = OFF_ZF + 65536;      // zero-init through here
constexpr size_t OFF_G1    = OFF_ZERO_END;           // fp32 [64][2048]
constexpr size_t OFF_G2    = OFF_G1 + 524288;        // fp32 [64][2048]
constexpr size_t OFF_Q     = OFF_G2 + 524288;        // fp32 [64][512]
constexpr size_t OFF_SC    = OFF_Q  + 131072;        // fp32 [64][1024] scores/attn
constexpr size_t OFF_BG1   = OFF_SC + 262144;        // fp32 [2048] bih1+bhh1
constexpr size_t OFF_BG2   = OFF_BG1 + 8192;         // fp32 [2048]
constexpr size_t OFF_W1    = OFF_BG2 + 8192;         // f16 [2048][1536]
constexpr size_t OFF_W2    = OFF_W1 + 6291456;       // f16 [2048][1024]
constexpr size_t OFF_WQ    = OFF_W2 + 4194304;       // f16 [512][512]
constexpr size_t OFF_WC    = OFF_WQ + 524288;        // f16 [512][1024]
constexpr size_t OFF_WE    = OFF_WC + 1048576;       // f16 [256][512]
constexpr size_t OFF_K     = OFF_WE + 262144;        // f16 [64*1024][512]  (67 MB)
constexpr size_t OFF_V     = OFF_K + 67108864;       // f16 [64*1024][512]  (67 MB)
constexpr size_t WS_NEED   = OFF_V + 67108864;       // ~150 MB (K+V live in L2)

// ---------------- WMMA fragment loaders (contiguous k = 16*half + e, same for A and B) ----------------
// A 16x32 (MxK): lane m=l&15, half=l>>4; elem e -> K = half*16+e
DEV v16h loadA(const _Float16* A, int lda, int m0, int k0, int lane) {
  int m = m0 + (lane & 15), half = lane >> 4;
  const _Float16* base = A + (size_t)m * lda + k0;
  union { v16h v; float4 f[2]; } u;
  u.f[0] = *reinterpret_cast<const float4*>(base + half * 16);        // 8 halves
  u.f[1] = *reinterpret_cast<const float4*>(base + half * 16 + 8);   // 8 halves
  return u.v;
}
// B 32x16 (KxN) from row-major W[N][ldw] (B[k][n] = W[n][k]): elem e -> K = half*16+e
DEV v16h loadBT(const _Float16* Wm, int ldw, int n0, int k0, int lane) {
  int n = n0 + (lane & 15), half = lane >> 4;
  const float4* p = reinterpret_cast<const float4*>(Wm + (size_t)n * ldw + k0 + half * 16);
  union { v16h v; float4 f[2]; } u;
  u.f[0] = p[0]; u.f[1] = p[1];
  return u.v;
}
DEV v16h loadA_f32(const float* A, int lda, int m0, int k0, int lane) {
  int m = m0 + (lane & 15), half = lane >> 4;
  const float* p = A + (size_t)m * lda + k0;
  union { v16h v; _Float16 h[16]; } u;
#pragma unroll
  for (int e = 0; e < 8; ++e) {
    u.h[e]     = (_Float16)p[half * 16 + e];
    u.h[e + 8] = (_Float16)p[half * 16 + 8 + e];
  }
  return u.v;
}
DEV v16h loadBT_f32(const float* Wm, int ldw, int n0, int k0, int lane) {
  int n = n0 + (lane & 15), half = lane >> 4;
  const float* p = Wm + (size_t)n * ldw + k0 + half * 16;
  union { v16h v; _Float16 h[16]; } u;
#pragma unroll
  for (int e = 0; e < 16; ++e) u.h[e] = (_Float16)p[e];
  return u.v;
}
DEV v8f gemm_tile(const _Float16* A, int lda, const _Float16* Bw, int ldb,
                  int m0, int n0, int K, int lane) {
  v8f acc = {};
  for (int kk = 0; kk < K; kk += 32) {
    v16h a = loadA(A, lda, m0, kk, lane);
    v16h b = loadBT(Bw, ldb, n0, kk, lane);
    acc = __builtin_amdgcn_wmma_f32_16x16x32_f16(false, a, false, b, (short)0, acc, false, false);
  }
  return acc;
}

DEV float sigm(float x) { return 1.0f / (1.0f + __expf(-x)); }

// agent-scope grid barrier (all NBLK blocks resident)
DEV void gsync(unsigned* bar) {
  __syncthreads();
  if (threadIdx.x == 0) {
    unsigned* cnt = bar;
    unsigned* gen = bar + 32;  // separate cache line
    unsigned g = __hip_atomic_load(gen, __ATOMIC_ACQUIRE, __HIP_MEMORY_SCOPE_AGENT);
    __threadfence();
    unsigned prev = __hip_atomic_fetch_add(cnt, 1u, __ATOMIC_ACQ_REL, __HIP_MEMORY_SCOPE_AGENT);
    if (prev + 1 == gridDim.x) {
      __hip_atomic_store(cnt, 0u, __ATOMIC_RELAXED, __HIP_MEMORY_SCOPE_AGENT);
      __hip_atomic_fetch_add(gen, 1u, __ATOMIC_RELEASE, __HIP_MEMORY_SCOPE_AGENT);
    } else {
      while (__hip_atomic_load(gen, __ATOMIC_ACQUIRE, __HIP_MEMORY_SCOPE_AGENT) == g)
        __builtin_amdgcn_s_sleep(1);
    }
    __threadfence();
  }
  __syncthreads();
}

// TDM: stage the f16 embedding table (256 rows x 512 cols, 256 KB) into LDS.
// D# per CDNA5 ISA: 2D tile, data_size=2B,
// tile_dim0=512, tile_dim1=256, tensor_dim0_stride=512, type=2.
DEV void tdm_load_emb_to_lds(const void* gsrc) {
  uint64_t ga = (uint64_t)(uintptr_t)gsrc;
  unsigned ldsa = __builtin_amdgcn_groupstaticsize();  // dynamic LDS starts here
  v4u g0;
  g0.x = 1u;                                            // count=1 (valid), no gather
  g0.y = ldsa;                                          // lds_addr (bytes)
  g0.z = (unsigned)(ga & 0xFFFFFFFFu);                  // global_addr[31:0]
  g0.w = (unsigned)((ga >> 32) & 0x01FFFFFFu) | (2u << 30);  // addr[56:32] | type=2
  v8i g1;
  g1[0] = (1 << 16);        // workgroup_mask=0 (not in cluster), data_size=1 (2 B)
  g1[1] = (512 << 16);      // tensor_dim0[15:0]=512 ; atomic_barrier_addr=0
  g1[2] = (256 << 16);      // tensor_dim0[31:16]=0 | tensor_dim1[15:0]=256
  g1[3] = (512 << 16);      // tensor_dim1[31:16]=0 | tile_dim0=512
  g1[4] = 256;              // tile_dim1=256 | tile_dim2=0
  g1[5] = 512;              // tensor_dim0_stride[31:0]=512
  g1[6] = 0;                // stride0[47:32]=0 | tensor_dim1_stride[15:0]=0
  g1[7] = 0;                // tensor_dim1_stride[47:16]=0
  v4i g2 = {0, 0, 0, 0};    // 2D tile: groups 2/3 unused
  v4i g3 = {0, 0, 0, 0};
#if defined(__clang_major__) && __clang_major__ >= 23
  v8i g4 = {0, 0, 0, 0, 0, 0, 0, 0};
  __builtin_amdgcn_tensor_load_to_lds(g0, g1, g2, g3, g4, 0);
#else
  __builtin_amdgcn_tensor_load_to_lds(g0, g1, g2, g3, 0);
#endif
  __builtin_amdgcn_s_wait_tensorcnt(0);
}

// ---------------- kernel 0: zero state + barrier ----------------
__global__ void __launch_bounds__(256) init_kernel(char* ws) {
  unsigned* p = (unsigned*)ws;
  size_t n = OFF_ZERO_END / 4;
  for (size_t i = blockIdx.x * 256 + threadIdx.x; i < n; i += (size_t)gridDim.x * 256)
    p[i] = 0u;
}

// ---------------- kernel 1: fuse + convert weights to f16 ----------------
__global__ void __launch_bounds__(256) prep_weights(
    char* ws, const float* Wih1, const float* Whh1, const float* Wih2, const float* Whh2,
    const float* Wq, const float* Wc, const float* Wemb,
    const float* bih1, const float* bhh1, const float* bih2, const float* bhh2) {
  _Float16* W1 = (_Float16*)(ws + OFF_W1);
  _Float16* W2 = (_Float16*)(ws + OFF_W2);
  _Float16* WQ = (_Float16*)(ws + OFF_WQ);
  _Float16* WC = (_Float16*)(ws + OFF_WC);
  _Float16* WE = (_Float16*)(ws + OFF_WE);
  float* BG1 = (float*)(ws + OFF_BG1);
  float* BG2 = (float*)(ws + OFF_BG2);
  const long N1 = 2048L * 1536, N2 = 2048L * 1024, N3 = 512L * 512,
             N4 = 512L * 1024, N5 = 256L * 512;
  const long total = N1 + N2 + N3 + N4 + N5 + 2048 + 2048;
  for (long i = (long)blockIdx.x * 256 + threadIdx.x; i < total; i += (long)gridDim.x * 256) {
    long r = i;
    if (r < N1) { int n = r / 1536, k = r % 1536;
      W1[r] = (_Float16)((k < 1024) ? Wih1[(long)n * 1024 + k] : Whh1[(long)n * 512 + k - 1024]);
      continue; }
    r -= N1;
    if (r < N2) { int n = r / 1024, k = r % 1024;
      W2[r] = (_Float16)((k < 512) ? Wih2[(long)n * 512 + k] : Whh2[(long)n * 512 + k - 512]);
      continue; }
    r -= N2;
    if (r < N3) { WQ[r] = (_Float16)Wq[r]; continue; }
    r -= N3;
    if (r < N4) { WC[r] = (_Float16)Wc[r]; continue; }
    r -= N4;
    if (r < N5) { WE[r] = (_Float16)Wemb[r]; continue; }
    r -= N5;
    if (r < 2048) { BG1[r] = bih1[r] + bhh1[r]; continue; }
    r -= 2048;
    BG2[r] = bih2[r] + bhh2[r];
  }
}

// ---------------- kernel 2: K/V precompute (WMMA, fp32->f16 on the fly) ----------
// K = enc @ Wk^T + bk, V = enc @ Wv^T + bv ; M = 64*1024 rows, N = 512, K = 512
__global__ void __launch_bounds__(256) kv_kernel(char* ws, const float* enc,
                                                 const float* Wk, const float* bk,
                                                 const float* Wv, const float* bv) {
  _Float16* Kf = (_Float16*)(ws + OFF_K);
  _Float16* Vf = (_Float16*)(ws + OFF_V);
  int lane = threadIdx.x & 31;
  long wid = (long)blockIdx.x * 8 + (threadIdx.x >> 5);
  long nw  = (long)gridDim.x * 8;
  const long ntiles = 4096L * 64;  // 4096 m-tiles x (32 K-ntiles + 32 V-ntiles)
  for (long tile = wid; tile < ntiles; tile += nw) {
    int nsel = (int)(tile & 63);
    int m0 = (int)(tile >> 6) * 16;
    bool isK = nsel < 32;
    int n0 = (nsel & 31) * 16;
    const float* Wm = isK ? Wk : Wv;
    v8f acc = {};
    for (int kk = 0; kk < 512; kk += 32) {
      v16h a = loadA_f32(enc, 512, m0, kk, lane);
      v16h b = loadBT_f32(Wm, 512, n0, kk, lane);
      acc = __builtin_amdgcn_wmma_f32_16x16x32_f16(false, a, false, b, (short)0, acc, false, false);
    }
    int half = lane >> 4, n = n0 + (lane & 15);
    float bias = (isK ? bk : bv)[n];
    _Float16* dst = isK ? Kf : Vf;
#pragma unroll
    for (int r = 0; r < 8; ++r) {
      long row = m0 + half * 8 + r;
      dst[row * 512 + n] = (_Float16)(acc[r] + bias);
    }
  }
}

// ---------------- kernel 3: persistent 400-step decoder ----------------
__global__ void __launch_bounds__(NTHR, 1) decoder_kernel(char* ws, const int* y,
                                                          const float* bq, const float* bc,
                                                          const float* bcls, float* out) {
  extern __shared__ _Float16 sWE[];   // 256 KB dynamic LDS: f16 Wemb [256][512]

  unsigned* bar = (unsigned*)(ws + OFF_BAR);
  float* h1  = (float*)(ws + OFF_H1);
  float* c1  = (float*)(ws + OFF_C1);
  float* h2  = (float*)(ws + OFF_H2);
  float* c2  = (float*)(ws + OFF_C2);
  float* ctx = (float*)(ws + OFF_CTX);
  _Float16* xcat1 = (_Float16*)(ws + OFF_XCAT1);
  _Float16* xcat2 = (_Float16*)(ws + OFF_XCAT2);
  _Float16* zin   = (_Float16*)(ws + OFF_ZIN);
  _Float16* zf    = (_Float16*)(ws + OFF_ZF);
  float* g1 = (float*)(ws + OFF_G1);
  float* g2 = (float*)(ws + OFF_G2);
  float* q  = (float*)(ws + OFF_Q);
  float* sc = (float*)(ws + OFF_SC);
  const float* BG1 = (const float*)(ws + OFF_BG1);
  const float* BG2 = (const float*)(ws + OFF_BG2);
  const _Float16* W1 = (const _Float16*)(ws + OFF_W1);
  const _Float16* W2 = (const _Float16*)(ws + OFF_W2);
  const _Float16* WQ = (const _Float16*)(ws + OFF_WQ);
  const _Float16* WC = (const _Float16*)(ws + OFF_WC);
  const _Float16* Kf = (const _Float16*)(ws + OFF_K);
  const _Float16* Vf = (const _Float16*)(ws + OFF_V);

  // TDM DMA: one wave per block pulls the embedding table into LDS, waits on
  // TENSORcnt, then the workgroup barrier releases the other 7 waves.
  if (threadIdx.x < 32) tdm_load_emb_to_lds((const void*)(ws + OFF_WE));
  __syncthreads();

  const int tid  = blockIdx.x * NTHR + threadIdx.x;     // 0..16383
  const int lane = threadIdx.x & 31;
  const int W    = blockIdx.x * 8 + (threadIdx.x >> 5); // 0..511
  const float scale = 0.04419417382415922f;             // 1/sqrt(512)

  for (int t = 0; t < 400; ++t) {
    // -- prep: xcat1 = [emb(prev_tok) (from LDS) | f16(ctx) | f16(h1)]
    for (int i = tid; i < 64 * 1536; i += NTID) {
      int b = i / 1536, col = i - b * 1536;
      _Float16 v;
      if (col < 512) {
        int tok = (t == 0) ? 1 : y[b * 400 + t - 1];   // SOS = 1
        v = sWE[tok * 512 + col];                      // ds_load from TDM-staged LDS
      } else if (col < 1024) v = (_Float16)ctx[b * 512 + col - 512];
      else                   v = (_Float16)h1[b * 512 + col - 1024];
      xcat1[i] = v;
    }
    gsync(bar);

    // -- LSTM1 gates: g1 = xcat1 @ [Wih1|Whh1]^T + bih1 + bhh1   (512 tiles, 1/wave)
    { int m0 = (W >> 7) * 16, n0 = (W & 127) * 16;
      v8f acc = gemm_tile(xcat1, 1536, W1, 1536, m0, n0, 1536, lane);
      int half = lane >> 4, n = n0 + (lane & 15);
      float bg = BG1[n];
#pragma unroll
      for (int r = 0; r < 8; ++r) g1[(m0 + half * 8 + r) * 2048 + n] = acc[r] + bg;
    }
    gsync(bar);

    // -- LSTM1 cell
    for (int i = tid; i < 64 * 512; i += NTID) {
      int b = i >> 9, u = i & 511;
      const float* g = g1 + b * 2048;
      float c = sigm(g[512 + u]) * c1[i] + sigm(g[u]) * tanhf(g[1024 + u]);
      float h = sigm(g[1536 + u]) * tanhf(c);
      c1[i] = c; h1[i] = h;
      xcat2[b * 1024 + u] = (_Float16)h;   // [h1 | h2_prev]
    }
    gsync(bar);

    // -- LSTM2 gates: g2 = [h1|h2] @ [Wih2|Whh2]^T + bias
    { int m0 = (W >> 7) * 16, n0 = (W & 127) * 16;
      v8f acc = gemm_tile(xcat2, 1024, W2, 1024, m0, n0, 1024, lane);
      int half = lane >> 4, n = n0 + (lane & 15);
      float bg = BG2[n];
#pragma unroll
      for (int r = 0; r < 8; ++r) g2[(m0 + half * 8 + r) * 2048 + n] = acc[r] + bg;
    }
    gsync(bar);

    // -- LSTM2 cell
    for (int i = tid; i < 64 * 512; i += NTID) {
      int b = i >> 9, u = i & 511;
      const float* g = g2 + b * 2048;
      float c = sigm(g[512 + u]) * c2[i] + sigm(g[u]) * tanhf(g[1024 + u]);
      float h = sigm(g[1536 + u]) * tanhf(c);
      c2[i] = c; h2[i] = h;
      zin[b * 1024 + u]         = (_Float16)h;  // [h2 | ctx] (ctx filled later)
      xcat2[b * 1024 + 512 + u] = (_Float16)h;  // h2 for next step's LSTM2
    }
    gsync(bar);

    // -- q = h2 @ Wq^T + bq   (128 tiles)
    if (W < 128) {
      int m0 = (W >> 5) * 16, n0 = (W & 31) * 16;
      v8f acc = gemm_tile(zin, 1024, WQ, 512, m0, n0, 512, lane);
      int half = lane >> 4, n = n0 + (lane & 15);
      float b_ = bq[n];
#pragma unroll
      for (int r = 0; r < 8; ++r) q[(m0 + half * 8 + r) * 512 + n] = acc[r] + b_;
    }
    gsync(bar);

    // -- attention scores (L2-resident f16 K; 128 dots/wave)
#pragma unroll 1
    for (int j = 0; j < 4; ++j) {
      int idx = W * 128 + j * 32 + lane;      // 0..65535
      int b = idx >> 10, s = idx & 1023;
      const _Float16* kr = Kf + (long)(b * 1024 + s) * 512;
      const float* qr = q + b * 512;
      __builtin_prefetch((const void*)(kr + 512), 0, 1);  // global_prefetch_b8
      float sum = 0.f;
#pragma unroll 4
      for (int h = 0; h < 512; h += 8) {
#pragma unroll
        for (int e = 0; e < 8; ++e) sum += qr[h + e] * (float)kr[h + e];
      }
      sc[idx] = sum * scale;
    }
    gsync(bar);

    // -- softmax per batch row (64 waves)
    if (W < 64) {
      float* row = sc + W * 1024;
      float vbuf[32];
      float mx = -1e30f;
#pragma unroll
      for (int i = 0; i < 32; ++i) { float v = row[lane + 32 * i]; vbuf[i] = v; mx = fmaxf(mx, v); }
      for (int off = 16; off; off >>= 1) mx = fmaxf(mx, __shfl_xor(mx, off, 32));
      float sum = 0.f;
#pragma unroll
      for (int i = 0; i < 32; ++i) { float e = __expf(vbuf[i] - mx); vbuf[i] = e; sum += e; }
      for (int off = 16; off; off >>= 1) sum += __shfl_xor(sum, off, 32);
      float inv = 1.0f / sum;
#pragma unroll
      for (int i = 0; i < 32; ++i) row[lane + 32 * i] = vbuf[i] * inv;
    }
    gsync(bar);

    // -- ctx = attn @ V  (L2-resident f16 V)
    for (int i = tid; i < 64 * 512; i += NTID) {
      int b = i >> 9, h = i & 511;
      const float* at = sc + b * 1024;
      const _Float16* vr = Vf + (long)b * 1024 * 512 + h;
      __builtin_prefetch((const void*)vr, 0, 1);
      float s = 0.f;
#pragma unroll 4
      for (int sI = 0; sI < 1024; ++sI) s += at[sI] * (float)vr[(long)sI * 512];
      ctx[i] = s;
      zin[b * 1024 + 512 + h] = (_Float16)s;
    }
    gsync(bar);

    // -- z = relu([h2|ctx] @ Wc^T + bc)  (128 tiles) -> f16
    if (W < 128) {
      int m0 = (W >> 5) * 16, n0 = (W & 31) * 16;
      v8f acc = gemm_tile(zin, 1024, WC, 1024, m0, n0, 1024, lane);
      int half = lane >> 4, n = n0 + (lane & 15);
      float b_ = bc[n];
#pragma unroll
      for (int r = 0; r < 8; ++r)
        zf[(m0 + half * 8 + r) * 512 + n] = (_Float16)fmaxf(acc[r] + b_, 0.f);
    }
    gsync(bar);

    // -- logits = z @ Wemb^T + b_cls  (64 tiles), B panel from TDM-staged LDS
    if (W < 64) {
      int m0 = (W >> 4) * 16, n0 = (W & 15) * 16;
      v8f acc = gemm_tile(zf, 512, sWE, 512, m0, n0, 512, lane);  // ds_load_b128 B
      int half = lane >> 4, n = n0 + (lane & 15);
      float b_ = bcls[n];
#pragma unroll
      for (int r = 0; r < 8; ++r) {
        long row = m0 + half * 8 + r;                 // batch index
        out[(row * 400 + t) * 256 + n] = acc[r] + b_;
      }
    }
    gsync(bar);

    // -- greedy argmax -> decodes at d_out[B*T*V + b*T + t]
    if (W < 64) {
      const float* lg = out + ((long)W * 400 + t) * 256;
      float best = -1e30f; int bi = 0;
#pragma unroll
      for (int j = 0; j < 8; ++j) {
        int idx = lane * 8 + j;
        float v = lg[idx];
        if (v > best) { best = v; bi = idx; }
      }
      for (int off = 16; off; off >>= 1) {
        float ov = __shfl_xor(best, off, 32);
        int   oi = __shfl_xor(bi, off, 32);
        if (ov > best || (ov == best && oi < bi)) { best = ov; bi = oi; }
      }
      if (lane == 0) out[64L * 400 * 256 + (long)W * 400 + t] = (float)bi;
    }
    gsync(bar);
  }
}

// ---------------- host launcher ----------------
extern "C" void kernel_launch(void* const* d_in, const int* in_sizes, int n_in,
                              void* d_out, int out_size, void* d_ws, size_t ws_size,
                              hipStream_t stream) {
  (void)in_sizes; (void)n_in; (void)out_size;
  if (ws_size < WS_NEED) return;  // need ~150 MB scratch (K/V resident in 192 MB L2)
  const float* enc  = (const float*)d_in[0];
  const int*   y    = (const int*)  d_in[1];
  const float* Wemb = (const float*)d_in[2];
  const float* W1ih = (const float*)d_in[3];
  const float* W1hh = (const float*)d_in[4];
  const float* b1ih = (const float*)d_in[5];
  const float* b1hh = (const float*)d_in[6];
  const float* W2ih = (const float*)d_in[7];
  const float* W2hh = (const float*)d_in[8];
  const float* b2ih = (const float*)d_in[9];
  const float* b2hh = (const float*)d_in[10];
  const float* Wq   = (const float*)d_in[11];
  const float* bq   = (const float*)d_in[12];
  const float* Wk   = (const float*)d_in[13];
  const float* bk   = (const float*)d_in[14];
  const float* Wv   = (const float*)d_in[15];
  const float* bv   = (const float*)d_in[16];
  const float* Wc   = (const float*)d_in[17];
  const float* bc   = (const float*)d_in[18];
  const float* bcls = (const float*)d_in[19];
  char* ws = (char*)d_ws;
  float* out = (float*)d_out;

  init_kernel<<<dim3(256), dim3(256), 0, stream>>>(ws);
  prep_weights<<<dim3(2048), dim3(256), 0, stream>>>(ws, W1ih, W1hh, W2ih, W2hh,
                                                     Wq, Wc, Wemb, b1ih, b1hh, b2ih, b2hh);
  kv_kernel<<<dim3(2048), dim3(256), 0, stream>>>(ws, enc, Wk, bk, Wv, bv);
  // 256 KB dynamic LDS for the TDM-staged embedding table
  decoder_kernel<<<dim3(NBLK), dim3(NTHR), 262144, stream>>>(ws, y, bq, bc, bcls, out);
}
